// DownLayer_32478542692842
// MI455X (gfx1250) — matrix operand order repaired and checked by reading, not compiled, not run on hardware
//
#include <hip/hip_runtime.h>
#include <hip/hip_bf16.h>

typedef __bf16 bf16;
typedef __attribute__((ext_vector_type(16))) __bf16 v16bf;
typedef __attribute__((ext_vector_type(8)))  float  v8f;
typedef __attribute__((ext_vector_type(4)))  int    v4i;

#define EPSF    1e-6f
#define LN_EPSF 1e-5f

#define B_    8
#define N_    4096
#define N0_   16384
#define CIN   256
#define COUT  512
#define HIN   64
#define WIN   64
#define HOUT  32
#define WOUT  32
#define KTOT  (CIN*9)   /* 2304 */
#define NGRID 256
#define NSAMP 768
#define NS    1024
#define NADA  (N_-NGRID) /* 3840 */

#if defined(__HIP_DEVICE_COMPILE__) && defined(__gfx1250__) && \
    __has_builtin(__builtin_amdgcn_global_load_async_to_lds_b128) && \
    __has_builtin(__builtin_amdgcn_s_wait_asynccnt)
#define USE_ASYNC 1
#else
#define USE_ASYNC 0
#endif

union AFrag { v16bf v; bf16 e[16]; };
union CFrag { v8f  v; float e[8]; };

// ---------------------------------------------------------------- utilities
__global__ void k_zero(float* __restrict__ p, long long n) {
  long long i = (long long)blockIdx.x * blockDim.x + threadIdx.x;
  if (i < n) p[i] = 0.f;
}

__global__ void k_wconv(const float* __restrict__ w, bf16* __restrict__ wb, int n) {
  int i = blockIdx.x * blockDim.x + threadIdx.x;
  if (i < n) wb[i] = (bf16)w[i];
}

// ------------------------------------------------------------- token2map scatter
__global__ void k_token2map(const float* __restrict__ x, const float* __restrict__ pos_orig,
                            const int* __restrict__ idx_agg,
                            float* __restrict__ msum, float* __restrict__ cnt) {
  long long t = (long long)blockIdx.x * blockDim.x + threadIdx.x;
  if (t >= (long long)B_ * N0_ * CIN) return;
  int c  = (int)(t & (CIN - 1));
  int bn = (int)(t >> 8);         // b*N0 + n0
  int b  = bn >> 14;
  float lx = fminf(fmaxf(pos_orig[(size_t)bn * 2 + 0], -1.f), 1.f);
  float ly = fminf(fmaxf(pos_orig[(size_t)bn * 2 + 1], -1.f), 1.f);
  int px = (int)rintf(0.5f * (lx + 1.f) * WIN - 0.5f);
  int py = (int)rintf(0.5f * (ly + 1.f) * HIN - 0.5f);
  px = min(max(px, 0), WIN - 1);
  py = min(max(py, 0), HIN - 1);
  int hw  = py * WIN + px;
  int src = idx_agg[bn];
  float v = x[((size_t)(b << 12) + src) * CIN + c];
  atomicAdd(&msum[((size_t)(b << 12) + hw) * CIN + c], v);
  if (c == 0) atomicAdd(&cnt[(b << 12) + hw], 1.f);
}

// normalize by count and convert to bf16, layout [B][CIN][H][W]
__global__ void k_norm2bf(const float* __restrict__ msum, const float* __restrict__ cnt,
                          bf16* __restrict__ fmapbf) {
  int t = blockIdx.x * blockDim.x + threadIdx.x;
  if (t >= B_ * CIN * HIN * WIN) return;
  int hw = t & 4095;
  int bc = t >> 12;
  int c  = bc & (CIN - 1);
  int b  = bc >> 8;
  float v = msum[((size_t)(b << 12) + hw) * CIN + c] / (cnt[(b << 12) + hw] + EPSF);
  fmapbf[t] = (bf16)v;
}

// -------------------------------------------------------- conv 3x3 s2 via WMMA
// Implicit im2col GEMM: M=8192 pixels, N=512 outch, K=2304.
// Block tile 128(M) x 64(N); 8 waves 4x2; each wave owns a 32x32 macro-tile
// = 2x2 WMMA accumulators (fragment reuse). Double-buffered LDS; weight tile
// staged with GLOBAL_LOAD_ASYNC_TO_LDS_B128 when available (ASYNCcnt pipeline).
__global__ __launch_bounds__(256) void k_conv_wmma(const bf16* __restrict__ fmapbf,
                                                   const bf16* __restrict__ wbf,
                                                   const float* __restrict__ convb,
                                                   float* __restrict__ out) {
  __shared__ __align__(16) bf16 As[2][128][40];   // 80B row stride: 16B aligned, bank-conflict-free
  __shared__ __align__(16) bf16 Bs[2][64][40];
  const int tid   = threadIdx.x;
  const int lane  = tid & 31, wid = tid >> 5;
  const int waveM = wid >> 1, waveN = wid & 1;    // 4 x 2 waves
  const int half  = lane >> 4, mn = lane & 15;
  const int mBase = blockIdx.x * 128;
  const int nBase = blockIdx.y * 64;

  // each thread stages 16 A elements; same kk, fixed set of 16 pixels -> hoist coords
  const int kk    = tid & 31;
  const int pbase = tid >> 5;
  int pb[16], poy[16], pox[16];
#pragma unroll
  for (int i = 0; i < 16; i++) {
    int gm = mBase + pbase + 8 * i;
    pb[i]  = gm >> 10;
    int rem = gm & 1023;
    poy[i] = rem >> 5;
    pox[i] = rem & 31;
  }

  auto loadA = [&](int buf, int k0) {
    int k  = k0 + kk;
    int ci = k / 9, r9 = k - 9 * ci;
    int kh = r9 / 3, kw = r9 - 3 * kh;
#pragma unroll
    for (int i = 0; i < 16; i++) {
      int iy = poy[i] * 2 - 1 + kh, ix = pox[i] * 2 - 1 + kw;
      bf16 bv = (bf16)0.f;
      if (iy >= 0 && iy < HIN && ix >= 0 && ix < WIN)
        bv = fmapbf[((pb[i] * CIN + ci) << 12) + (iy << 6) + ix];
      As[buf][pbase + 8 * i][kk] = bv;
    }
  };
  auto loadB = [&](int buf, int k0) {
#if USE_ASYNC
    int row = tid >> 2, ch4 = tid & 3;            // 64 rows x 4 x 16B chunks
    const bf16* src = wbf + (size_t)(nBase + row) * KTOT + k0 + ch4 * 8;
    __builtin_amdgcn_global_load_async_to_lds_b128(
        (__attribute__((address_space(1))) v4i*)src,
        (__attribute__((address_space(3))) v4i*)&Bs[buf][row][ch4 * 8],
        0, 0);
#else
    for (int e = tid; e < 64 * 32; e += 256)
      Bs[buf][e >> 5][e & 31] = wbf[(size_t)(nBase + (e >> 5)) * KTOT + k0 + (e & 31)];
#endif
  };

  CFrag acc[2][2];
#pragma unroll
  for (int i = 0; i < 2; i++)
#pragma unroll
    for (int j = 0; j < 2; j++)
#pragma unroll
      for (int r = 0; r < 8; r++) acc[i][j].e[r] = 0.f;

  loadB(0, 0);
  loadA(0, 0);
#if USE_ASYNC
  __builtin_amdgcn_s_wait_asynccnt(0);
#endif
  __syncthreads();

  int buf = 0;
  for (int k0 = 0; k0 < KTOT; k0 += 32, buf ^= 1) {
    if (k0 + 32 < KTOT) {        // prefetch next chunk into other buffer
      loadB(buf ^ 1, k0 + 32);
      loadA(buf ^ 1, k0 + 32);
    }
    // ---- pack fragments per ISA 16-bit wave32 layouts
    AFrag a[2], bm[2];
#pragma unroll
    for (int s = 0; s < 2; s++) {
      int rowA = waveM * 32 + s * 16 + mn;
      int rowB = waveN * 32 + s * 16 + mn;
#pragma unroll
      for (int r = 0; r < 8; r++) {
        int kbA = (r < 4) ? ((half ? 8 : 0) + 2 * r) : ((half ? 24 : 16) + 2 * (r - 4));
        a[s].e[2 * r]     = As[buf][rowA][kbA];
        a[s].e[2 * r + 1] = As[buf][rowA][kbA + 1];
        int kbB = (r < 4) ? ((half ? 8 : 0) + 2 * r) : ((half ? 24 : 16) + 2 * (r - 4));
        bm[s].e[2 * r]     = Bs[buf][rowB][kbB];
        bm[s].e[2 * r + 1] = Bs[buf][rowB][kbB + 1];
      }
    }
    acc[0][0].v = __builtin_amdgcn_wmma_f32_16x16x32_bf16(false, a[0].v, false, bm[0].v,
                                                          (short)0, acc[0][0].v, false, false);
    acc[0][1].v = __builtin_amdgcn_wmma_f32_16x16x32_bf16(false, a[0].v, false, bm[1].v,
                                                          (short)0, acc[0][1].v, false, false);
    acc[1][0].v = __builtin_amdgcn_wmma_f32_16x16x32_bf16(false, a[1].v, false, bm[0].v,
                                                          (short)0, acc[1][0].v, false, false);
    acc[1][1].v = __builtin_amdgcn_wmma_f32_16x16x32_bf16(false, a[1].v, false, bm[1].v,
                                                          (short)0, acc[1][1].v, false, false);
#if USE_ASYNC
    __builtin_amdgcn_s_wait_asynccnt(0);
#endif
    __syncthreads();
  }

#pragma unroll
  for (int j = 0; j < 2; j++) {
    const int ch = nBase + waveN * 32 + j * 16 + mn;
    const float bias = convb[ch];
#pragma unroll
    for (int i = 0; i < 2; i++) {
#pragma unroll
      for (int r = 0; r < 8; r++) {
        int gm = mBase + waveM * 32 + i * 16 + r + 8 * half;  // D: VGPR r -> M=r (+8 upper half)
        out[(size_t)gm * COUT + ch] = acc[i][j].e[r] + bias;
      }
    }
  }
}

// -------------------------------------------------------------- map2token
__global__ void k_map2token(const float* __restrict__ convout, const float* __restrict__ pos_orig,
                            const int* __restrict__ idx_agg, const float* __restrict__ aggw,
                            float* __restrict__ tsum, float* __restrict__ wsum) {
  long long t = (long long)blockIdx.x * blockDim.x + threadIdx.x;
  if (t >= (long long)B_ * N0_ * COUT) return;
  int c  = (int)(t & (COUT - 1));
  int bn = (int)(t >> 9);
  int b  = bn >> 14;
  float lx = fminf(fmaxf(pos_orig[(size_t)bn * 2 + 0], -1.f), 1.f);
  float ly = fminf(fmaxf(pos_orig[(size_t)bn * 2 + 1], -1.f), 1.f);
  float gx = 0.5f * (lx + 1.f) * WOUT - 0.5f;
  float gy = 0.5f * (ly + 1.f) * HOUT - 0.5f;
  int xl = min(max((int)floorf(gx), 0), WOUT - 1);
  int xh = min(xl + 1, WOUT - 1);
  float xw = fminf((float)xh, gx) - (float)xl;
  int yl = min(max((int)floorf(gy), 0), HOUT - 1);
  int yh = min(yl + 1, HOUT - 1);
  float yw = fminf((float)yh, gy) - (float)yl;
  const float* fm = convout + (size_t)b * HOUT * WOUT * COUT;
  float v = fm[(size_t)(yl * WOUT + xl) * COUT + c] * (1.f - xw) * (1.f - yw)
          + fm[(size_t)(yl * WOUT + xh) * COUT + c] * xw * (1.f - yw)
          + fm[(size_t)(yh * WOUT + xl) * COUT + c] * (1.f - xw) * yw
          + fm[(size_t)(yh * WOUT + xh) * COUT + c] * xw * yw;
  float wv = aggw[bn];
  int dst  = idx_agg[bn];
  atomicAdd(&tsum[((size_t)(b << 12) + dst) * COUT + c], v * wv);
  if (c == 0) atomicAdd(&wsum[(b << 12) + dst], wv);
}

// ---------------------------------------- finalize x + LayerNorm + conf + exp
__global__ __launch_bounds__(256) void k_ln_conf(float* __restrict__ tsum, const float* __restrict__ wsum,
                                                 const float* __restrict__ lng, const float* __restrict__ lnb,
                                                 const float* __restrict__ cw, const float* __restrict__ cb,
                                                 float* __restrict__ conf, float* __restrict__ wexp) {
  int tok = blockIdx.x;
  int tid = threadIdx.x;
  __shared__ float r1[256], r2[256];
  float wsv = wsum[tok] + EPSF;
  float v0 = tsum[(size_t)tok * COUT + tid] / wsv;
  float v1 = tsum[(size_t)tok * COUT + tid + 256] / wsv;
  tsum[(size_t)tok * COUT + tid]       = v0;     // x (reused later for merge)
  tsum[(size_t)tok * COUT + tid + 256] = v1;
  r1[tid] = v0 + v1;
  r2[tid] = v0 * v0 + v1 * v1;
  __syncthreads();
  for (int s = 128; s > 0; s >>= 1) {
    if (tid < s) { r1[tid] += r1[tid + s]; r2[tid] += r2[tid + s]; }
    __syncthreads();
  }
  float mean = r1[0] / (float)COUT;
  float var  = r2[0] / (float)COUT - mean * mean;
  float rstd = rsqrtf(var + LN_EPSF);
  float xn0 = (v0 - mean) * rstd * lng[tid]       + lnb[tid];
  float xn1 = (v1 - mean) * rstd * lng[tid + 256] + lnb[tid + 256];
  __syncthreads();
  r1[tid] = xn0 * cw[tid] + xn1 * cw[tid + 256];
  __syncthreads();
  for (int s = 128; s > 0; s >>= 1) {
    if (tid < s) r1[tid] += r1[tid + s];
    __syncthreads();
  }
  if (tid == 0) {
    float c = r1[0] + cb[0];
    conf[tok] = c;
    wexp[tok] = expf(c);
  }
}

// ------------------------------------ per-batch top-k via bitonic sort in LDS
__global__ __launch_bounds__(1024) void k_topk(const float* __restrict__ conf,
                                               const float* __restrict__ gu,
                                               int* __restrict__ idown) {
  __shared__ float key[4096];
  __shared__ int   kid[4096];
  int b = blockIdx.x, tid = threadIdx.x;
  for (int j = tid; j < 4096; j += 1024) {
    if (j < NADA) {
      float u = gu[b * NADA + j];
      float g = -logf(-logf(u + 1e-6f) + 1e-6f);
      key[j]  = conf[(b << 12) + NGRID + j] + g;
    } else {
      key[j] = -3.0e38f;
    }
    kid[j] = j;
  }
  __syncthreads();
  for (int k = 2; k <= 4096; k <<= 1) {
    for (int j = k >> 1; j > 0; j >>= 1) {
      for (int t = tid; t < 2048; t += 1024) {
        int i   = (t / j) * (j << 1) + (t % j);
        int ixj = i + j;
        bool dirDesc = ((i & k) == 0);
        float a = key[i], c = key[ixj];
        if ((a < c) == dirDesc) {
          key[i] = c; key[ixj] = a;
          int ti = kid[i]; kid[i] = kid[ixj]; kid[ixj] = ti;
        }
      }
      __syncthreads();
    }
  }
  for (int j = tid; j < NSAMP; j += 1024) idown[b * NSAMP + j] = kid[j];
}

__global__ __launch_bounds__(1024) void k_posdown(const float* __restrict__ pos,
                                                  const int* __restrict__ idown,
                                                  float* __restrict__ o_pd) {
  int b = blockIdx.x, i = threadIdx.x;                 // i in [0,1024)
  int src = (i < NGRID) ? i : (NGRID + idown[b * NSAMP + (i - NGRID)]);
  o_pd[((size_t)((b << 10) + i)) * 2 + 0] = pos[((size_t)((b << 12) + src)) * 2 + 0];
  o_pd[((size_t)((b << 10) + i)) * 2 + 1] = pos[((size_t)((b << 12) + src)) * 2 + 1];
}

// --------------------------------------------- nearest-centroid + weight sums
__global__ __launch_bounds__(256) void k_argmin(const float* __restrict__ pos,
                                                const float* __restrict__ pd,
                                                const float* __restrict__ wexp,
                                                int* __restrict__ idxmin, float* __restrict__ aw) {
  __shared__ float sp[NS * 2];
  int b = blockIdx.x;
  int n = blockIdx.y * 256 + threadIdx.x;
  for (int i = threadIdx.x; i < NS * 2; i += 256) sp[i] = pd[(size_t)b * NS * 2 + i];
  __syncthreads();
  float px = pos[((size_t)((b << 12) + n)) * 2 + 0];
  float py = pos[((size_t)((b << 12) + n)) * 2 + 1];
  int best = 0; float bd = 3.0e38f;
  for (int s = 0; s < NS; s++) {
    float dx = px - sp[2 * s], dy = py - sp[2 * s + 1];
    float d = dx * dx + dy * dy;
    if (d < bd) { bd = d; best = s; }
  }
  idxmin[(b << 12) + n] = best;
  atomicAdd(&aw[(b << 10) + best], wexp[(b << 12) + n]);
}

__global__ void k_nw(const float* __restrict__ wexp, const float* __restrict__ aw,
                     const int* __restrict__ idxmin, float* __restrict__ nw) {
  int t = blockIdx.x * blockDim.x + threadIdx.x;
  if (t >= B_ * N_) return;
  int b = t >> 12;
  nw[t] = wexp[t] / (aw[(b << 10) + idxmin[t]] + EPSF);
}

__global__ __launch_bounds__(256) void k_xdown(const float* __restrict__ xtok,
                                               const float* __restrict__ nw,
                                               const int* __restrict__ idxmin,
                                               float* __restrict__ xdown) {
  int tok = blockIdx.x, tid = threadIdx.x;
  int b = tok >> 12;
  float w = nw[tok];
  int dst = (b << 10) + idxmin[tok];
  for (int c = tid; c < COUT; c += 256)
    atomicAdd(&xdown[(size_t)dst * COUT + c], xtok[(size_t)tok * COUT + c] * w);
}

__global__ void k_down_idx(const int* __restrict__ idx_agg, const int* __restrict__ idxmin,
                           const float* __restrict__ nw, const float* __restrict__ aggw,
                           float* __restrict__ o_idx, float* __restrict__ o_agg,
                           unsigned* __restrict__ maxw) {
  int t = blockIdx.x * blockDim.x + threadIdx.x;
  if (t >= B_ * N0_) return;
  int b = t >> 14;
  int i = idx_agg[t];
  int tok = (b << 12) + i;
  float wt  = nw[tok];
  float awd = aggw[t] * wt;
  o_idx[t] = (float)idxmin[tok];
  o_agg[t] = awd;
  atomicMax(&maxw[b], __float_as_uint(awd));   // positive floats: uint order == float order
}

__global__ void k_agg_norm(float* __restrict__ o_agg, const unsigned* __restrict__ maxw) {
  int t = blockIdx.x * blockDim.x + threadIdx.x;
  if (t >= B_ * N0_) return;
  int b = t >> 14;
  o_agg[t] /= __uint_as_float(maxw[b]);
}

// ---------------------------------------------------------------------- launch
extern "C" void kernel_launch(void* const* d_in, const int* in_sizes, int n_in,
                              void* d_out, int out_size, void* d_ws, size_t ws_size,
                              hipStream_t stream) {
  (void)in_sizes; (void)n_in; (void)out_size; (void)ws_size;
  const float* x          = (const float*)d_in[0];
  const float* pos        = (const float*)d_in[1];
  const float* pos_orig   = (const float*)d_in[2];
  const int*   idx_agg    = (const int*)  d_in[3];
  const float* agg_weight = (const float*)d_in[4];
  const float* gumbel_u   = (const float*)d_in[5];
  const float* conv_w     = (const float*)d_in[6];
  const float* conv_b     = (const float*)d_in[7];
  const float* ln_g       = (const float*)d_in[8];
  const float* ln_b       = (const float*)d_in[9];
  const float* conf_w     = (const float*)d_in[10];
  const float* conf_b     = (const float*)d_in[11];

  float* W = (float*)d_ws;
  // accumulators (contiguous, zeroed each call); all offsets 16B aligned
  const size_t SZ_MSUM = (size_t)B_ * HIN * WIN * CIN;   // 8,388,608
  const size_t SZ_CNT  = (size_t)B_ * HIN * WIN;         // 32,768
  const size_t SZ_TSUM = (size_t)B_ * N_ * COUT;         // 16,777,216
  const size_t SZ_WSUM = (size_t)B_ * N_;
  const size_t SZ_AW   = (size_t)B_ * NS;
  const size_t SZ_MAXW = 64;
  size_t off = 0;
  float* msum   = W + off; off += SZ_MSUM;
  float* cnt    = W + off; off += SZ_CNT;
  float* tsum   = W + off; off += SZ_TSUM;               // becomes x after k_ln_conf
  float* wsum   = W + off; off += SZ_WSUM;
  float* aw     = W + off; off += SZ_AW;
  unsigned* maxw = (unsigned*)(W + off); off += SZ_MAXW;
  const size_t nAccum = off;
  // scratch written-before-read each call
  float* conf    = W + off; off += (size_t)B_ * N_;
  float* wexp    = W + off; off += (size_t)B_ * N_;
  float* nw      = W + off; off += (size_t)B_ * N_;
  float* convout = W + off; off += (size_t)B_ * HOUT * WOUT * COUT;
  int*   idxmin  = (int*)(W + off); off += (size_t)B_ * N_;
  int*   idown   = (int*)(W + off); off += (size_t)B_ * NSAMP;
  bf16*  wbf     = (bf16*)(W + off); off += ((size_t)COUT * KTOT + 1) / 2;   // 16B aligned
  bf16*  fmapbf  = (bf16*)(W + off); off += ((size_t)B_ * CIN * HIN * WIN + 1) / 2;

  float* o_xdown = (float*)d_out;                                    // 8*1024*512
  float* o_pd    = o_xdown + (size_t)B_ * NS * COUT;                 // 8*1024*2
  float* o_idx   = o_pd + (size_t)B_ * NS * 2;                       // 8*16384
  float* o_agg   = o_idx + (size_t)B_ * N0_;                         // 8*16384

  const int T = 256;
  // zero accumulators + x_down output region
  k_zero<<<(unsigned)((nAccum + T - 1) / T), T, 0, stream>>>(W, (long long)nAccum);
  k_zero<<<(unsigned)(((size_t)B_ * NS * COUT + T - 1) / T), T, 0, stream>>>(o_xdown, (long long)B_ * NS * COUT);
  // weights -> bf16
  k_wconv<<<(COUT * KTOT + T - 1) / T, T, 0, stream>>>(conv_w, wbf, COUT * KTOT);
  // token2map
  k_token2map<<<(unsigned)(((size_t)B_ * N0_ * CIN) / T), T, 0, stream>>>(x, pos_orig, idx_agg, msum, cnt);
  k_norm2bf<<<(B_ * CIN * HIN * WIN) / T, T, 0, stream>>>(msum, cnt, fmapbf);
  // conv via WMMA: grid (M/128, N/64)
  k_conv_wmma<<<dim3((B_ * HOUT * WOUT) / 128, COUT / 64), T, 0, stream>>>(fmapbf, wbf, conv_b, convout);
  // map2token
  k_map2token<<<(unsigned)(((size_t)B_ * N0_ * COUT) / T), T, 0, stream>>>(convout, pos_orig, idx_agg, agg_weight, tsum, wsum);
  // finalize + LN + conf
  k_ln_conf<<<B_ * N_, T, 0, stream>>>(tsum, wsum, ln_g, ln_b, conf_w, conf_b, conf, wexp);
  // gumbel top-k per batch
  k_topk<<<B_, 1024, 0, stream>>>(conf, gumbel_u, idown);
  k_posdown<<<B_, 1024, 0, stream>>>(pos, idown, o_pd);
  // merge tokens
  k_argmin<<<dim3(B_, N_ / 256), T, 0, stream>>>(pos, o_pd, wexp, idxmin, aw);
  k_nw<<<(B_ * N_ + T - 1) / T, T, 0, stream>>>(wexp, aw, idxmin, nw);
  k_xdown<<<B_ * N_, T, 0, stream>>>(tsum, nw, idxmin, o_xdown);
  k_down_idx<<<(B_ * N0_ + T - 1) / T, T, 0, stream>>>(idx_agg, idxmin, nw, agg_weight, o_idx, o_agg, maxw);
  k_agg_norm<<<(B_ * N0_ + T - 1) / T, T, 0, stream>>>(o_agg, maxw);
}
